// ResRnn_32762010533969
// MI455X (gfx1250) — matrix operand order repaired and checked by reading, not compiled, not run on hardware
//
#include <hip/hip_runtime.h>
#include <hip/hip_bf16.h>

// ---------------------------------------------------------------------------
// ResRNN scan for MI455X (gfx1250).
//   state  s: (256, 1024), recurrence independent per batch row.
//   step:  s' = [x_t | s[:, :960]];  h = |s' W1^T + b1|;
//          s  = 0.97 s' + 0.03 (h W2^T + b2)
// Strategy: 16 persistent workgroups x 16 batch rows (no inter-WG traffic).
// State lives in LDS (bf16, WMMA-A-ready, row-major 16x1024). Weights packed
// to bf16 once per launch and streamed from L2 (2 MB each -> L2 resident).
// Matrix math via v_wmma_f32_16x16x32_bf16 (wave32).
// 512 threads/WG = 16 waves = 4 waves/SIMD -> ~256 VGPRs/wave, so the 2-deep
// software pipeline (operands for k+1 loaded before WMMAs of k) fits with NO
// scratch spills. Each wave owns 4 N-tiles: one LDS A-load pair feeds 4 WMMAs.
// ---------------------------------------------------------------------------

typedef __attribute__((ext_vector_type(16))) __bf16 v16bf;
typedef __attribute__((ext_vector_type(8)))  float  v8f;

#define STREAM   1024
#define BATCH    256
#define SEQ      1024
#define INSZ     64
#define BT       16          // batch rows per workgroup
#define NWG      (BATCH / BT)
#define KSTEPS   (STREAM / 32)
#define NTHREADS 512
#define TILES    4           // N-tiles per wave (64 tiles / 16 waves)
#define P_ID     0.97f
#define Q_ID     0.03f

__device__ __forceinline__ v8f wmma_bf16(v16bf a, v16bf b, v8f c) {
    // D = A(16x32) * B(32x16) + C ; emits v_wmma_f32_16x16x32_bf16
    return __builtin_amdgcn_wmma_f32_16x16x32_bf16(
        /*neg_a=*/false, a, /*neg_b=*/false, b,
        /*c_mod=*/(short)0, c, /*reuse_a=*/false, /*reuse_b=*/false);
}

union AOp { uint4 q[2]; v16bf v; };

// One wave computes four 16x16 output tiles of  A(16xK) * W^T.
// Abuf: LDS, bf16 row-major [16][1024]  (A operand, lane M = l%16).
// Wp  : global bf16 row-major [1024][1024] (B operand, lane N = l%16,
//       16 contiguous K values per lane half). 2-deep software pipeline.
__device__ __forceinline__ void gemm_4tiles(const __bf16* __restrict__ Abuf,
                                            const __bf16* __restrict__ Wp,
                                            int lm, int hi, const int* __restrict__ j,
                                            v8f acc[TILES]) {
    const __bf16* aRow = Abuf + lm * STREAM;
    const __bf16* wRow[TILES];
#pragma unroll
    for (int i = 0; i < TILES; ++i)
        wRow[i] = Wp + (size_t)j[i] * STREAM + 16 * hi;

    AOp A[2], B[TILES][2];

    // ---- prologue: load iteration 0 operands into buffer 0 ----
    A[0].q[0] = *(const uint4*)(aRow + 16 * hi);
    A[0].q[1] = *(const uint4*)(aRow + 8 + 16 * hi);
#pragma unroll
    for (int i = 0; i < TILES; ++i) {
        const uint4* w = (const uint4*)(wRow[i]);
        B[i][0].q[0] = w[0]; B[i][0].q[1] = w[1];
    }

#pragma unroll 4
    for (int kt = 0; kt < KSTEPS; ++kt) {
        const int cur = kt & 1;
        const int nxt = cur ^ 1;
        // ---- issue next iteration's loads BEFORE consuming current ones ----
        if (kt + 1 < KSTEPS) {
            const int kn = (kt + 1) * 32;
            A[nxt].q[0] = *(const uint4*)(aRow + kn + 16 * hi);
            A[nxt].q[1] = *(const uint4*)(aRow + kn + 8 + 16 * hi);
#pragma unroll
            for (int i = 0; i < TILES; ++i) {
                const uint4* w = (const uint4*)(wRow[i] + kn);
                B[i][nxt].q[0] = w[0]; B[i][nxt].q[1] = w[1];
                // one cacheline-granular prefetch burst every 4th K-step
                if ((kt & 3) == 0)
                    __builtin_prefetch((const void*)(w + 64), 0, 0);  // +1 KB
            }
        }
        // ---- consume operands loaded one iteration ago ----
#pragma unroll
        for (int i = 0; i < TILES; ++i)
            acc[i] = wmma_bf16(A[cur].v, B[i][cur].v, acc[i]);
    }
}

__global__ void pack_weights_bf16(const float* __restrict__ W,
                                  __bf16* __restrict__ Wp, int n) {
    int i = blockIdx.x * blockDim.x + threadIdx.x;
    if (i < n) Wp[i] = (__bf16)W[i];
}

__global__ __launch_bounds__(NTHREADS, 1)
void resrnn_scan_kernel(const float*  __restrict__ input,   // (SEQ,BATCH,64)
                        const __bf16* __restrict__ W1p,     // (1024,1024) bf16
                        const float*  __restrict__ b1,
                        const __bf16* __restrict__ W2p,
                        const float*  __restrict__ b2,
                        float* __restrict__ out) {           // (BATCH,64)
    extern __shared__ __align__(32) char smem[];
    __bf16* s_bf = (__bf16*)smem;                 // 16 x 1024 state (32 KB)
    __bf16* h_bf = (__bf16*)(smem + BT * STREAM * sizeof(__bf16)); // 32 KB

    const int tid    = threadIdx.x;
    const int batch0 = blockIdx.x * BT;
    const int lm = tid & 15;          // lane % 16
    const int hi = (tid >> 4) & 1;    // lane / 16
    const int wv = tid >> 5;          // wave id 0..15

    int j[TILES];
    float b1j[TILES], b2j[TILES];
#pragma unroll
    for (int i = 0; i < TILES; ++i) {
        j[i]   = (TILES * wv + i) * 16 + lm;
        b1j[i] = b1[j[i]];
        b2j[i] = b2[j[i]];
    }

    // zero initial state
    for (int i = tid; i < BT * STREAM; i += NTHREADS) s_bf[i] = (__bf16)0.0f;
    __syncthreads();

    for (int t = 0; t < SEQ; ++t) {
        // ---- shift-in x_t (cols 0..63); cols 64.. already hold s[0:960] ----
#pragma unroll
        for (int i = tid; i < BT * INSZ; i += NTHREADS) {
            const int xm = i >> 6, xc = i & 63;
            s_bf[xm * STREAM + xc] =
                (__bf16)input[((size_t)t * BATCH + batch0 + xm) * INSZ + xc];
        }
        __syncthreads();

        // ---- GEMM1: h = |s W1^T + b1| -> h_bf ----
        v8f acc[TILES] = {};
        gemm_4tiles(s_bf, W1p, lm, hi, j, acc);
#pragma unroll
        for (int i = 0; i < TILES; ++i)
#pragma unroll
            for (int r = 0; r < 8; ++r) {
                const int M = r + 8 * hi;        // C/D layout: M = vgpr + 8*(l/16)
                h_bf[M * STREAM + j[i]] = (__bf16)fabsf(acc[i][r] + b1j[i]);
            }
        __syncthreads();

        // ---- GEMM2 + residual blend ----
        v8f acc2[TILES] = {};
        gemm_4tiles(h_bf, W2p, lm, hi, j, acc2);

        float sn[TILES][8];
#pragma unroll
        for (int i = 0; i < TILES; ++i)
#pragma unroll
            for (int r = 0; r < 8; ++r) {
                const int M = r + 8 * hi;
                sn[i][r] = P_ID * (float)s_bf[M * STREAM + j[i]]
                         + Q_ID * (acc2[i][r] + b2j[i]);
            }
        __syncthreads();   // all residual reads of s_bf done before shifting

        // ---- shifted write-back: col j -> col j+64 ; last 64 cols of final
        //      step are the kernel output ----
#pragma unroll
        for (int i = 0; i < TILES; ++i)
#pragma unroll
            for (int r = 0; r < 8; ++r) {
                const int M = r + 8 * hi;
                if (j[i] < STREAM - INSZ)
                    s_bf[M * STREAM + j[i] + INSZ] = (__bf16)sn[i][r];
                else if (t == SEQ - 1)
                    out[(size_t)(batch0 + M) * INSZ + (j[i] - (STREAM - INSZ))]
                        = sn[i][r];
            }
        __syncthreads();
    }
}

extern "C" void kernel_launch(void* const* d_in, const int* in_sizes, int n_in,
                              void* d_out, int out_size, void* d_ws, size_t ws_size,
                              hipStream_t stream) {
    const float* input = (const float*)d_in[0];
    const float* W1    = (const float*)d_in[1];
    const float* b1    = (const float*)d_in[2];
    const float* W2    = (const float*)d_in[3];
    const float* b2    = (const float*)d_in[4];
    float* out = (float*)d_out;

    __bf16* W1p = (__bf16*)d_ws;
    __bf16* W2p = W1p + (size_t)STREAM * STREAM;

    const int nW = STREAM * STREAM;
    pack_weights_bf16<<<(nW + 255) / 256, 256, 0, stream>>>(W1, W1p, nW);
    pack_weights_bf16<<<(nW + 255) / 256, 256, 0, stream>>>(W2, W2p, nW);

    const size_t shmem = 2u * BT * STREAM * sizeof(__bf16);   // 64 KB
    resrnn_scan_kernel<<<NWG, NTHREADS, shmem, stream>>>(input, W1p, b1, W2p, b2, out);
}
